// ConvGRU_50156628083068
// MI455X (gfx1250) — matrix operand order repaired and checked by reading, not compiled, not run on hardware
//
#include <hip/hip_runtime.h>
#include <hip/hip_bf16.h>

// ---------------------------------------------------------------------------
// 2-layer GRU (B=32, T=512, D=H=512) as a persistent cooperative kernel.
// bf16 WMMA (v_wmma_f32_16x16x32_bf16) with f32 accumulation.
// 32 workgroups x 128 threads; each WG owns 16 output columns and keeps its
// 3-gate weight slice (96 KB bf16) in LDS for the whole layer.
// ---------------------------------------------------------------------------

typedef __attribute__((ext_vector_type(16))) __bf16 v16bf;
typedef __attribute__((ext_vector_type(8)))  float  v8f;

#define GRU_B   32
#define GRU_T   512
#define GRU_H   512
#define GRU_K   1024          // D + H
#define GRU_L   2
#define NWG     32            // 32 WGs x 16 cols = 512 outputs
#define NTHR    128           // 4 waves per WG

// ----- bf16 bit helpers (storage is unsigned short everywhere) -------------
__device__ inline unsigned short f2bf(float f) {
  unsigned u = __float_as_uint(f);
  unsigned r = ((u >> 16) & 1u) + 0x7FFFu;   // round-to-nearest-even
  return (unsigned short)((u + r) >> 16);
}
__device__ inline float bf2f(unsigned short h) {
  return __uint_as_float(((unsigned)h) << 16);
}
// Fast activations: single v_rcp_f32 instead of the precise-division sequence
// (v_div_scale/v_div_fmas/v_div_fixup). ~1 ulp rcp error << bf16 noise.
__device__ inline float sigm(float x) {
  return __builtin_amdgcn_rcpf(1.0f + __expf(-x));
}
__device__ inline float tanh_fast(float x) {
  float xc = fminf(fmaxf(x, -12.0f), 12.0f);
  float e  = __expf(2.0f * xc);
  return (e - 1.0f) * __builtin_amdgcn_rcpf(e + 1.0f);
}

// ----- agent-scope grid barrier (32 co-resident WGs) ------------------------
__device__ inline void grid_sync(unsigned* counter, unsigned* gen) {
  __syncthreads();
  if (threadIdx.x == 0) {
    unsigned g0 = __hip_atomic_load(gen, __ATOMIC_RELAXED, __HIP_MEMORY_SCOPE_AGENT);
    __threadfence();
    unsigned arrived = __hip_atomic_fetch_add(counter, 1u, __ATOMIC_ACQ_REL,
                                              __HIP_MEMORY_SCOPE_AGENT);
    if (arrived == NWG - 1) {
      __hip_atomic_store(counter, 0u, __ATOMIC_RELAXED, __HIP_MEMORY_SCOPE_AGENT);
      __hip_atomic_fetch_add(gen, 1u, __ATOMIC_RELEASE, __HIP_MEMORY_SCOPE_AGENT);
    } else {
      while (__hip_atomic_load(gen, __ATOMIC_ACQUIRE, __HIP_MEMORY_SCOPE_AGENT) == g0) {
        __builtin_amdgcn_s_sleep(1);
      }
    }
    __threadfence();
  }
  __syncthreads();
}

// ----- one 16x16 output tile, K=1024, via 32 chained WMMAs ------------------
// The k range is statically split at k=512 (x-part vs h-part) so the inner
// loops carry no pointer selects: addresses fold into immediate offsets.
// xrow : bf16 row of x-part      (index k,      k in [0,512))
// hrow : bf16 row of h / (h*r)   (index k-512,  k in [512,1024))
// wrow : bf16 LDS weight row W[j][*] (row-major, contiguous in k)
__device__ __attribute__((always_inline)) inline
v8f gemm_tile(const unsigned short* __restrict__ xrow,
              const unsigned short* __restrict__ hrow,
              const unsigned short* __restrict__ wrow,
              int half, float bias) {
  v8f acc = {bias, bias, bias, bias, bias, bias, bias, bias};
  // x half: k in [0, 512)
#pragma unroll 8
  for (int kb = 0; kb < GRU_H; kb += 32) {
    union { v16bf v; uint4 q[2]; } ua, ub;
    const unsigned short* s1 = xrow + kb + half * 8;       // elems 0..7
    ua.q[0] = *(const uint4*)s1;
    ua.q[1] = *(const uint4*)(s1 + 16);                    // elems 8..15
    const unsigned short* ws = wrow + kb + half * 8;       // same k order as A
    ub.q[0] = *(const uint4*)ws;
    ub.q[1] = *(const uint4*)(ws + 16);
    acc = __builtin_amdgcn_wmma_f32_16x16x32_bf16(
        false, ua.v, false, ub.v, (short)0, acc, false, false);
  }
  // h half: k in [512, 1024), hrow indexed from 0, weights offset by 512
#pragma unroll 8
  for (int kb = 0; kb < GRU_H; kb += 32) {
    union { v16bf v; uint4 q[2]; } ua, ub;
    const unsigned short* s1 = hrow + kb + half * 8;
    ua.q[0] = *(const uint4*)s1;
    ua.q[1] = *(const uint4*)(s1 + 16);
    const unsigned short* ws = wrow + GRU_H + kb + half * 8;
    ub.q[0] = *(const uint4*)ws;
    ub.q[1] = *(const uint4*)(ws + 16);
    acc = __builtin_amdgcn_wmma_f32_16x16x32_bf16(
        false, ua.v, false, ub.v, (short)0, acc, false, false);
  }
  return acc;
}

// ----- prep kernels ---------------------------------------------------------
// Wb layout: [l][g][j][k] bf16, g: 0=update(Wu), 1=reset(Wr), 2=out(Wo)
__global__ __launch_bounds__(256) void prep_weights(
    const float* __restrict__ Wu, const float* __restrict__ Wr,
    const float* __restrict__ Wo, unsigned short* __restrict__ Wb) {
  const int total = GRU_L * 3 * GRU_H * GRU_K;
  int idx = blockIdx.x * blockDim.x + threadIdx.x;
  if (idx >= total) return;
  int k  = idx & (GRU_K - 1);
  int r1 = idx >> 10;
  int j  = r1 & (GRU_H - 1);
  int r2 = r1 >> 9;
  int g  = r2 % 3;
  int l  = r2 / 3;
  const float* W = (g == 0) ? Wu : (g == 1) ? Wr : Wo;   // [l][j][k]
  Wb[idx] = f2bf(W[((size_t)(l * GRU_H + j)) * GRU_K + k]);
}

// x (B,T,D) f32  ->  act0 [t][b][d] bf16
__global__ __launch_bounds__(256) void prep_x(
    const float* __restrict__ x, unsigned short* __restrict__ act0) {
  const int total = GRU_B * GRU_T * GRU_H;
  int idx = blockIdx.x * blockDim.x + threadIdx.x;
  if (idx >= total) return;
  int d = idx & (GRU_H - 1);
  int r = idx >> 9;
  int t = r & (GRU_T - 1);
  int b = r >> 9;
  act0[((size_t)(t * GRU_B + b)) * GRU_H + d] = f2bf(x[idx]);
}

__global__ void prep_sync(unsigned* sync) {
  if (threadIdx.x == 0 && blockIdx.x == 0) { sync[0] = 0u; sync[1] = 0u; }
}

// ----- main persistent GRU kernel ------------------------------------------
__global__ __launch_bounds__(NTHR, 1) void gru_main(
    const unsigned short* __restrict__ Wb,
    unsigned short* __restrict__ act0,      // layer0 input  [t][b][512] bf16
    unsigned short* __restrict__ act1,      // layer0 output [t][b][512] bf16
    unsigned short* __restrict__ hbuf,      // [2][32][512] bf16 ping-pong h
    unsigned short* __restrict__ hr,        // [32][512] bf16 (h * r)
    float* __restrict__ ug,                 // [32][512] f32 update gate
    const float* __restrict__ br, const float* __restrict__ bu,
    const float* __restrict__ bo,
    float* __restrict__ out,                // (L,B,H) f32
    unsigned* __restrict__ sync) {
  __shared__ unsigned short lds_w[3 * 16 * GRU_K];   // 96 KB: WG's weight slice

  const int tid  = threadIdx.x;
  const int wg   = blockIdx.x;
  const int wv   = tid >> 5;
  const int lane = tid & 31;
  const int n    = lane & 15;      // D/C column within tile, and A row within tile
  const int half = lane >> 4;
  const int j    = wg * 16 + n;    // global output column this lane produces

  for (int l = 0; l < GRU_L; ++l) {
    // Stage this WG's 3-gate weight slice into LDS (once per layer, 96 KB).
    for (int idx = tid; idx < 3 * 16 * GRU_K; idx += NTHR) {
      int k  = idx & (GRU_K - 1);
      int jj = (idx >> 10) & 15;
      int g  = idx >> 14;
      lds_w[idx] =
          Wb[((size_t)((l * 3 + g) * GRU_H + wg * 16 + jj)) * GRU_K + k];
    }
    // h0 = 0 for this WG's column slice (ping buffer 0).
    for (int idx = tid; idx < GRU_B * 16; idx += NTHR) {
      int b  = idx >> 4;
      int nn = idx & 15;
      hbuf[(size_t)b * GRU_H + wg * 16 + nn] = 0;
    }
    grid_sync(sync, sync + 1);

    const unsigned short* actin  = (l == 0) ? act0 : act1;
    unsigned short*       actout = (l == 0) ? act1 : act0;  // l==1 seq unused
    int ping = 0;

    for (int t = 0; t < GRU_T; ++t) {
      // ---- phase 1: update (g=0) & reset (g=1) gates, 4 waves ----
      {
        const int g  = (wv < 2) ? 0 : 1;
        const int mt = wv & 1;
        const int m  = mt * 16 + n;  // A row for this lane
        const unsigned short* xrow = actin + (size_t)(t * GRU_B + m) * GRU_H;
        const unsigned short* hrow = hbuf + (size_t)(ping * GRU_B + m) * GRU_H;
        const unsigned short* wrow = lds_w + (g * 16 + n) * GRU_K;
        const float bias = ((g == 0) ? bu : br)[l * GRU_H + j];
        v8f acc = gemm_tile(xrow, hrow, wrow, half, bias);
        if (g == 0) {
#pragma unroll
          for (int e = 0; e < 8; ++e) {
            int b = mt * 16 + half * 8 + e;
            ug[(size_t)b * GRU_H + j] = sigm(acc[e]);
          }
        } else {
#pragma unroll
          for (int e = 0; e < 8; ++e) {
            int b = mt * 16 + half * 8 + e;
            float rv = sigm(acc[e]);
            float hv = bf2f(hbuf[(size_t)(ping * GRU_B + b) * GRU_H + j]);
            hr[(size_t)b * GRU_H + j] = f2bf(hv * rv);
          }
        }
      }
      grid_sync(sync, sync + 1);

      // ---- phase 2: output gate + state update, 2 waves ----
      if (wv < 2) {
        const int mt = wv;
        const int m  = mt * 16 + n;
        const unsigned short* xrow = actin + (size_t)(t * GRU_B + m) * GRU_H;
        const unsigned short* hrow = hr + (size_t)m * GRU_H;
        const unsigned short* wrow = lds_w + (2 * 16 + n) * GRU_K;
        const float bias = bo[l * GRU_H + j];
        v8f acc = gemm_tile(xrow, hrow, wrow, half, bias);
#pragma unroll
        for (int e = 0; e < 8; ++e) {
          int b = mt * 16 + half * 8 + e;
          float ov = tanh_fast(acc[e]);
          float hv = bf2f(hbuf[(size_t)(ping * GRU_B + b) * GRU_H + j]);
          float uv = ug[(size_t)b * GRU_H + j];
          float hn = hv * (1.0f - uv) + ov * uv;
          unsigned short hb = f2bf(hn);
          hbuf[(size_t)((1 - ping) * GRU_B + b) * GRU_H + j] = hb;
          actout[(size_t)(t * GRU_B + b) * GRU_H + j] = hb;
          if (t == GRU_T - 1) out[(size_t)(l * GRU_B + b) * GRU_H + j] = hn;
        }
      }
      grid_sync(sync, sync + 1);
      ping ^= 1;
    }
  }
}

// ---------------------------------------------------------------------------
extern "C" void kernel_launch(void* const* d_in, const int* in_sizes, int n_in,
                              void* d_out, int out_size, void* d_ws, size_t ws_size,
                              hipStream_t stream) {
  (void)in_sizes; (void)n_in; (void)out_size; (void)ws_size;
  const float* x  = (const float*)d_in[0];
  const float* Wr = (const float*)d_in[1];
  const float* br = (const float*)d_in[2];
  const float* Wu = (const float*)d_in[3];
  const float* bu = (const float*)d_in[4];
  const float* Wo = (const float*)d_in[5];
  const float* bo = (const float*)d_in[6];
  float* out = (float*)d_out;

  // workspace carve-out (256B aligned)
  auto align256 = [](size_t v) { return (v + 255) & ~(size_t)255; };
  char* ws = (char*)d_ws;
  size_t off = 0;
  unsigned short* Wb = (unsigned short*)(ws + off);
  off += align256((size_t)GRU_L * 3 * GRU_H * GRU_K * 2);          // 6.29 MB
  unsigned short* act0 = (unsigned short*)(ws + off);
  off += align256((size_t)GRU_T * GRU_B * GRU_H * 2);              // 16 MB
  unsigned short* act1 = (unsigned short*)(ws + off);
  off += align256((size_t)GRU_T * GRU_B * GRU_H * 2);              // 16 MB
  unsigned short* hbuf = (unsigned short*)(ws + off);
  off += align256((size_t)2 * GRU_B * GRU_H * 2);
  unsigned short* hr = (unsigned short*)(ws + off);
  off += align256((size_t)GRU_B * GRU_H * 2);
  float* ug = (float*)(ws + off);
  off += align256((size_t)GRU_B * GRU_H * 4);
  unsigned* sync = (unsigned*)(ws + off);
  off += 256;

  const int wtot = GRU_L * 3 * GRU_H * GRU_K;
  const int xtot = GRU_B * GRU_T * GRU_H;
  prep_weights<<<(wtot + 255) / 256, 256, 0, stream>>>(Wu, Wr, Wo, Wb);
  prep_x<<<(xtot + 255) / 256, 256, 0, stream>>>(x, act0);
  prep_sync<<<1, 32, 0, stream>>>(sync);
  gru_main<<<NWG, NTHR, 0, stream>>>(Wb, act0, act1, hbuf, hr, ug,
                                     br, bu, bo, out, sync);
}
